// My_MultiHeadSelfAttention_20091857011102
// MI455X (gfx1250) — matrix operand order repaired and checked by reading, not compiled, not run on hardware
//
#include <hip/hip_runtime.h>
#include <hip/hip_bf16.h>
#include <math.h>

typedef __attribute__((ext_vector_type(16))) __bf16 v16bf;
typedef __attribute__((ext_vector_type(8)))  __bf16 v8bf;
typedef __attribute__((ext_vector_type(4)))  __bf16 v4bf;
typedef __attribute__((ext_vector_type(8)))  float  v8f;

#define D_MODEL 1024
#define NHEAD   16
#define DH      64
#define BATCH   2
#define SEQ     2048
#define MROWS   (BATCH * SEQ)   // 4096

// ---------------------------------------------------------------------------
// WMMA fragment helpers (wave32, 16x16x32 bf16, f32 accum)
// A-matrix 16x32 (MxK): lane L holds row L%16; K = {0..7,16..23} (L<16) or
// {8..15,24..31} (L>=16).
// B-matrix 32x16 (KxN): lane L holds column L%16; K = {0..7,16..23} (L<16) or
// {8..15,24..31} (L>=16), same order as A.
// C/D 16x16 f32: lane L holds col L%16, rows r + 8*(L/16), r = VGPR index.
// ---------------------------------------------------------------------------
__device__ __forceinline__ v16bf load_frag_a(const __bf16* p, int ld) {
    int lane = threadIdx.x & 31;
    const __bf16* r = p + (lane & 15) * ld + ((lane & 16) ? 8 : 0);
    v8bf lo = *(const v8bf*)(r);
    v8bf hi = *(const v8bf*)(r + 16);
    return __builtin_shufflevector(lo, hi, 0,1,2,3,4,5,6,7,8,9,10,11,12,13,14,15);
}

__device__ __forceinline__ v16bf load_frag_b(const __bf16* p, int ld) {
    int lane = threadIdx.x & 31;
    const __bf16* r = p + (lane & 15) * ld + ((lane & 16) ? 8 : 0);
    v8bf lo = *(const v8bf*)(r);
    v8bf hi = *(const v8bf*)(r + 16);
    return __builtin_shufflevector(lo, hi, 0,1,2,3,4,5,6,7,8,9,10,11,12,13,14,15);
}

// per-lane prefetch covering every row of a fragment (global_prefetch_b8)
__device__ __forceinline__ void prefetch_frag(const __bf16* p, int ld) {
    int lane = threadIdx.x & 31;
    __builtin_prefetch(p + (lane & 15) * ld + ((lane & 16) ? 16 : 0), 0, 3);
}

__device__ __forceinline__ v8f wmma_bf16(v16bf a, v16bf b, v8f c) {
    return __builtin_amdgcn_wmma_f32_16x16x32_bf16(
        /*neg_a=*/false, a, /*neg_b=*/false, b,
        /*c_mod=*/(short)0, c, /*reuse_a=*/false, /*reuse_b=*/false);
}

// ---------------------------------------------------------------------------
// f32 -> bf16 conversion (vectorized x4)
// ---------------------------------------------------------------------------
__global__ void cvt_f32_bf16(const float4* __restrict__ in,
                             __bf16* __restrict__ out, int n4) {
    int i = blockIdx.x * blockDim.x + threadIdx.x;
    if (i < n4) {
        float4 v = in[i];
        v4bf o;
        o[0] = (__bf16)v.x; o[1] = (__bf16)v.y;
        o[2] = (__bf16)v.z; o[3] = (__bf16)v.w;
        *(v4bf*)(out + 4 * (size_t)i) = o;
    }
}

// ---------------------------------------------------------------------------
// GEMM: Y[M,N] = A[M,K] @ W[N,K]^T + bias[N]
// One wave computes a 32x64 tile: 2 A-frags x 4 B-frags -> 8 WMMAs / k-step.
// No double-buffering (keeps live VGPRs ~120, no spills); all 6 fragment
// loads are issued as a group before the 8 WMMAs, prefetch runs one k-step
// ahead (speculative, so unconditional is safe).
// mode 0: store bf16 [B,H,T,DH]   (Q / K layout)
// mode 2: store bf16 [B,H,DH,T]   (V transposed)
// mode 3: store f32  [M,N]        (final projection -> d_out)
// ---------------------------------------------------------------------------
__global__ void __launch_bounds__(256)
gemm_wmma(const __bf16* __restrict__ A,
          const __bf16* __restrict__ W,
          const float*  __restrict__ bias,
          __bf16* __restrict__ outb,
          float*  __restrict__ outf,
          int mode) {
    int wave  = (int)((blockIdx.x * blockDim.x + threadIdx.x) >> 5);
    int lane  = threadIdx.x & 31;
    const int NT = D_MODEL / 64;     // 16 col-tiles of 64
    int m0 = (wave / NT) * 32;
    int n0 = (wave % NT) * 64;

    const __bf16* arow0 = A + (size_t)m0 * D_MODEL;
    const __bf16* arow1 = A + (size_t)(m0 + 16) * D_MODEL;
    const __bf16* wbase = W + (size_t)n0 * D_MODEL;

    v8f acc[2][4] = {};

    for (int k = 0; k < D_MODEL; k += 32) {
        // speculative prefetch one k-step ahead (drops silently at the end)
        prefetch_frag(arow0 + k + 32, D_MODEL);
        prefetch_frag(arow1 + k + 32, D_MODEL);
        prefetch_frag(wbase + (size_t)(0 * 16) * D_MODEL + k + 32, D_MODEL);
        prefetch_frag(wbase + (size_t)(1 * 16) * D_MODEL + k + 32, D_MODEL);
        prefetch_frag(wbase + (size_t)(2 * 16) * D_MODEL + k + 32, D_MODEL);
        prefetch_frag(wbase + (size_t)(3 * 16) * D_MODEL + k + 32, D_MODEL);

        v16bf a0 = load_frag_a(arow0 + k, D_MODEL);
        v16bf a1 = load_frag_a(arow1 + k, D_MODEL);
        v16bf b0 = load_frag_b(wbase + (size_t)(0 * 16) * D_MODEL + k, D_MODEL);
        v16bf b1 = load_frag_b(wbase + (size_t)(1 * 16) * D_MODEL + k, D_MODEL);
        v16bf b2 = load_frag_b(wbase + (size_t)(2 * 16) * D_MODEL + k, D_MODEL);
        v16bf b3 = load_frag_b(wbase + (size_t)(3 * 16) * D_MODEL + k, D_MODEL);

        acc[0][0] = wmma_bf16(a0, b0, acc[0][0]);
        acc[1][0] = wmma_bf16(a1, b0, acc[1][0]);
        acc[0][1] = wmma_bf16(a0, b1, acc[0][1]);
        acc[1][1] = wmma_bf16(a1, b1, acc[1][1]);
        acc[0][2] = wmma_bf16(a0, b2, acc[0][2]);
        acc[1][2] = wmma_bf16(a1, b2, acc[1][2]);
        acc[0][3] = wmma_bf16(a0, b3, acc[0][3]);
        acc[1][3] = wmma_bf16(a1, b3, acc[1][3]);
    }

    int colL  = lane & 15;
    int rbase = (lane >> 4) * 8;
#pragma unroll
    for (int rt = 0; rt < 2; ++rt) {
#pragma unroll
        for (int t = 0; t < 4; ++t) {
            int   n  = n0 + t * 16 + colL;
            float bv = bias[n];
#pragma unroll
            for (int r = 0; r < 8; ++r) {
                float v = acc[rt][t][r] + bv;
                int   m = m0 + rt * 16 + rbase + r;
                if (mode == 3) {
                    outf[(size_t)m * D_MODEL + n] = v;
                } else {
                    int b = m / SEQ, tt = m % SEQ, h = n / DH, d = n % DH;
                    size_t idx = (mode == 2)
                        ? ((((size_t)b * NHEAD + h) * DH + d) * SEQ + tt)   // Vt
                        : ((((size_t)b * NHEAD + h) * SEQ + tt) * DH + d);  // Q/K
                    outb[idx] = (__bf16)v;
                }
            }
        }
    }
}

// ---------------------------------------------------------------------------
// Flash attention (causal). One wave per (b, h, 16-row q tile).
// Q,K: [B,H,T,DH] bf16 ; Vt: [B,H,DH,T] bf16 ; att out: [B,T,D] bf16.
// ---------------------------------------------------------------------------
__global__ void __launch_bounds__(128)
flash_attn(const __bf16* __restrict__ Q,
           const __bf16* __restrict__ K,
           const __bf16* __restrict__ Vt,
           __bf16* __restrict__ att) {
    __shared__ alignas(64) __bf16 plds[4][16 * 32];  // per-wave P tile slab
    int lane = threadIdx.x & 31;
    int wib  = threadIdx.x >> 5;
    int wave = (int)((blockIdx.x * blockDim.x + threadIdx.x) >> 5);

    int qt = wave & 127;           // SEQ/16 = 128 q tiles
    int h  = (wave >> 7) & (NHEAD - 1);
    int b  = wave >> 11;

    const __bf16* Qh = Q  + (((size_t)b * NHEAD + h) * SEQ) * DH;
    const __bf16* Kh = K  + (((size_t)b * NHEAD + h) * SEQ) * DH;
    const __bf16* Vh = Vt + (((size_t)b * NHEAD + h) * DH) * SEQ;
    __bf16* lp = plds[wib];

    v16bf qa0 = load_frag_a(Qh + (size_t)qt * 16 * DH + 0,  DH);
    v16bf qa1 = load_frag_a(Qh + (size_t)qt * 16 * DH + 32, DH);

    float mrow[8], lrow[8];
#pragma unroll
    for (int r = 0; r < 8; ++r) { mrow[r] = -__builtin_inff(); lrow[r] = 0.f; }
    v8f o[4] = {};

    int   colL  = lane & 15;
    int   rbase = (lane >> 4) * 8;
    int   qrow0 = qt * 16;
    int   jmax  = (qrow0 + 15) / 32;     // inclusive; causal bound
    const float scale = 0.125f;          // 1/sqrt(64)

    for (int j = 0; j <= jmax; ++j) {
        int k0 = j * 32;
        // ---- scores: S = Q (16x64) . K^T (64x32), two 16x16 C fragments
        v8f s0 = {}, s1 = {};
        s0 = wmma_bf16(qa0, load_frag_b(Kh + (size_t)k0 * DH + 0,  DH), s0);
        s0 = wmma_bf16(qa1, load_frag_b(Kh + (size_t)k0 * DH + 32, DH), s0);
        s1 = wmma_bf16(qa0, load_frag_b(Kh + (size_t)(k0 + 16) * DH + 0,  DH), s1);
        s1 = wmma_bf16(qa1, load_frag_b(Kh + (size_t)(k0 + 16) * DH + 32, DH), s1);

        // ---- V fragments: independent of softmax; issue early so the heavy
        // VALU section below hides their latency before the P.V WMMAs.
        v16bf vb[4];
#pragma unroll
        for (int t = 0; t < 4; ++t)
            vb[t] = load_frag_b(Vh + (size_t)(t * 16) * SEQ + k0, SEQ);

        // speculative prefetch of next j-step K/V tiles (safe past the end)
        {
            int k1 = k0 + 32;
            prefetch_frag(Kh + (size_t)k1 * DH, DH);
            prefetch_frag(Kh + (size_t)(k1 + 16) * DH, DH);
            prefetch_frag(Vh + k1, SEQ);
            prefetch_frag(Vh + (size_t)32 * SEQ + k1, SEQ);
        }

        float alpha[8];
#pragma unroll
        for (int r = 0; r < 8; ++r) {
            int   q  = qrow0 + rbase + r;
            float v0 = s0[r] * scale;
            float v1 = s1[r] * scale;
            if (k0 + colL      > q) v0 = -__builtin_inff();   // causal mask
            if (k0 + 16 + colL > q) v1 = -__builtin_inff();
            // row max across the 16 lanes holding this row
            float t = fmaxf(v0, v1);
#pragma unroll
            for (int off = 1; off < 16; off <<= 1)
                t = fmaxf(t, __shfl_xor(t, off, 32));
            float mnew = fmaxf(mrow[r], t);
            alpha[r] = __expf(mrow[r] - mnew);
            float p0 = __expf(v0 - mnew);
            float p1 = __expf(v1 - mnew);
            float rs = p0 + p1;
#pragma unroll
            for (int off = 1; off < 16; off <<= 1)
                rs += __shfl_xor(rs, off, 32);
            lrow[r] = lrow[r] * alpha[r] + rs;
            mrow[r] = mnew;
            // stage P (bf16) in LDS to re-layout C-frag -> A-frag
            int row = rbase + r;
            lp[row * 32 + colL]      = (__bf16)p0;
            lp[row * 32 + 16 + colL] = (__bf16)p1;
        }
        // rescale running output
#pragma unroll
        for (int t = 0; t < 4; ++t)
#pragma unroll
            for (int r = 0; r < 8; ++r) o[t][r] *= alpha[r];

        v16bf pa = load_frag_a(lp, 32);   // P as 16x32 A fragment
#pragma unroll
        for (int t = 0; t < 4; ++t)
            o[t] = wmma_bf16(pa, vb[t], o[t]);
    }

    // epilogue: normalize and write concat-head output (bf16)
#pragma unroll
    for (int t = 0; t < 4; ++t) {
        int d = h * DH + t * 16 + colL;
#pragma unroll
        for (int r = 0; r < 8; ++r) {
            int q = qrow0 + rbase + r;
            att[((size_t)b * SEQ + q) * D_MODEL + d] = (__bf16)(o[t][r] / lrow[r]);
        }
    }
}

// ---------------------------------------------------------------------------
// Launch
// ---------------------------------------------------------------------------
extern "C" void kernel_launch(void* const* d_in, const int* in_sizes, int n_in,
                              void* d_out, int out_size, void* d_ws, size_t ws_size,
                              hipStream_t stream) {
    const float* x  = (const float*)d_in[0];
    // d_in[1] = mask (causal tril) -- applied analytically in-kernel
    const float* wq = (const float*)d_in[2];
    const float* bq = (const float*)d_in[3];
    const float* wk = (const float*)d_in[4];
    const float* bk = (const float*)d_in[5];
    const float* wv = (const float*)d_in[6];
    const float* bv = (const float*)d_in[7];
    const float* wo = (const float*)d_in[8];
    const float* bo = (const float*)d_in[9];
    float* out = (float*)d_out;

    char* ws = (char*)d_ws;
    const size_t actE = (size_t)MROWS * D_MODEL;       // 4M elems
    const size_t wE   = (size_t)D_MODEL * D_MODEL;     // 1M elems
    __bf16* xh   = (__bf16*)ws;  ws += actE * sizeof(__bf16);
    __bf16* wqh  = (__bf16*)ws;  ws += wE * sizeof(__bf16);
    __bf16* wkh  = (__bf16*)ws;  ws += wE * sizeof(__bf16);
    __bf16* wvh  = (__bf16*)ws;  ws += wE * sizeof(__bf16);
    __bf16* woh  = (__bf16*)ws;  ws += wE * sizeof(__bf16);
    __bf16* Qb   = (__bf16*)ws;  ws += actE * sizeof(__bf16);
    __bf16* Kb   = (__bf16*)ws;  ws += actE * sizeof(__bf16);
    __bf16* Vtb  = (__bf16*)ws;  ws += actE * sizeof(__bf16);
    __bf16* atth = (__bf16*)ws;  ws += actE * sizeof(__bf16);

    // 1) bf16 conversions
    {
        int n4 = (int)(actE / 4);
        cvt_f32_bf16<<<(n4 + 255) / 256, 256, 0, stream>>>((const float4*)x, xh, n4);
        int w4 = (int)(wE / 4);
        cvt_f32_bf16<<<(w4 + 255) / 256, 256, 0, stream>>>((const float4*)wq, wqh, w4);
        cvt_f32_bf16<<<(w4 + 255) / 256, 256, 0, stream>>>((const float4*)wk, wkh, w4);
        cvt_f32_bf16<<<(w4 + 255) / 256, 256, 0, stream>>>((const float4*)wv, wvh, w4);
        cvt_f32_bf16<<<(w4 + 255) / 256, 256, 0, stream>>>((const float4*)wo, woh, w4);
    }

    // 2) Q/K/V projections: (4096/32)*(1024/64) = 2048 waves = 256 blocks x 256
    const int gemmBlocks = (MROWS / 32) * (D_MODEL / 64) * 32 / 256; // 256
    gemm_wmma<<<gemmBlocks, 256, 0, stream>>>(xh, wqh, bq, Qb,  nullptr, 0);
    gemm_wmma<<<gemmBlocks, 256, 0, stream>>>(xh, wkh, bk, Kb,  nullptr, 0);
    gemm_wmma<<<gemmBlocks, 256, 0, stream>>>(xh, wvh, bv, Vtb, nullptr, 2);

    // 3) causal flash attention: 2*16*128 = 4096 waves = 1024 blocks x 128
    flash_attn<<<1024, 128, 0, stream>>>(Qb, Kb, Vtb, atth);

    // 4) output projection -> f32 d_out
    gemm_wmma<<<gemmBlocks, 256, 0, stream>>>(atth, woh, bo, nullptr, out, 3);
}
